// RelMultiHeadAttention_19439021981916
// MI455X (gfx1250) — matrix operand order repaired and checked by reading, not compiled, not run on hardware
//
#include <hip/hip_runtime.h>
#include <hip/hip_bf16.h>

// ---------------------------------------------------------------------------
// Types for CDNA5 WMMA (gfx1250, wave32)
// ---------------------------------------------------------------------------
typedef __attribute__((ext_vector_type(16))) __bf16 v16bf;
typedef __attribute__((ext_vector_type(8)))  float  v8f;

union BF16x16 {
    v16bf v;
    unsigned short u[16];
    uint4 q[2];
};

__device__ __forceinline__ unsigned short f2bf(float f) {
    unsigned int u = __float_as_uint(f);
    unsigned int r = u + 0x7FFFu + ((u >> 16) & 1u);   // round-to-nearest-even
    return (unsigned short)(r >> 16);
}

__device__ __forceinline__ v8f wmma_bf16(const v16bf& a, const v16bf& b, const v8f& c) {
    return __builtin_amdgcn_wmma_f32_16x16x32_bf16(false, a, false, b, (short)0, c, false, false);
}

// ---------------------------------------------------------------------------
// Problem constants
// ---------------------------------------------------------------------------
#define DMODEL 1024
#define NHEAD  16
#define DHEAD  64
#define BSZ    2
#define QLEN   1024
#define MLEN   1024
#define KLEN   2048

// ---------------------------------------------------------------------------
// fp32 -> bf16 conversion kernels
// ---------------------------------------------------------------------------
__global__ void k_f32_to_bf16(const float* __restrict__ s, unsigned short* __restrict__ d, int n) {
    int i = blockIdx.x * blockDim.x + threadIdx.x;
    if (i < n) d[i] = f2bf(s[i]);
}

// k_input = concat(mem, x) along seq, converted to bf16.  [B, KLEN, DMODEL]
__global__ void k_build_kin(const float* __restrict__ mem, const float* __restrict__ x,
                            unsigned short* __restrict__ kin) {
    int i = blockIdx.x * blockDim.x + threadIdx.x;
    const int total = BSZ * KLEN * DMODEL;
    if (i >= total) return;
    int e = i & (DMODEL - 1);
    int t = i >> 10;            // b*KLEN + k
    int k = t & (KLEN - 1);
    int b = t >> 11;
    float v = (k < MLEN) ? mem[((b << 10) + k) * DMODEL + e]
                         : x[((b << 10) + (k - MLEN)) * DMODEL + e];
    kin[i] = f2bf(v);
}

// ---------------------------------------------------------------------------
// WMMA GEMM:  Y[M x 1024] = X(bf16)[M x 1024] * W(bf16)[1024 x 1024]^T
//   outF != null : write fp32 row-major
//   outT != null : write bf16 TRANSPOSED per batch: VT[b][col][k],
//                  where row = b*KLEN + k  (used for the V projection)
//   else         : write bf16 to outA ((y+biasA)*sAB) and optionally
//                  outB ((y+biasB)*sAB)   [sAB is a power of two -> exact]
// Block = 256 threads (8 waves), wave computes a 16x64 tile.
// Inner K-loop is explicitly double-buffered (ping-pong fragments) so that
// ~10 b128 loads stay in flight behind each group of 4 WMMAs.
// grid = (1024/64, M/128)
// ---------------------------------------------------------------------------
__global__ __launch_bounds__(256) void k_gemm_xwT(
        const unsigned short* __restrict__ X, const unsigned short* __restrict__ W, int M,
        unsigned short* outA, const float* biasA,
        unsigned short* outB, const float* biasB,
        float sAB, float* outF, unsigned short* outT)
{
    const int lane = threadIdx.x & 31;
    const int wave = threadIdx.x >> 5;
    const int n0   = blockIdx.x * 64;
    const int m0   = blockIdx.y * 128 + wave * 16;
    const int nsub = lane & 15;
    const int half = lane >> 4;
    const int klo  = half * 8;    // (unused)
    const int kbh  = half * 16;   // B-frag K offset (one 16-elt chunk)

    v8f acc[4];
    #pragma unroll
    for (int t = 0; t < 4; ++t) acc[t] = (v8f){};

    const unsigned short* xrow  = X + (size_t)(m0 + nsub) * DMODEL;
    const unsigned short* wbase = W + (size_t)(n0 + nsub) * DMODEL + kbh;

    BF16x16 a0, a1, b0[4], b1[4];
    auto loadA = [&](BF16x16& a, int k0) {
        const unsigned short* ap = xrow + k0 + kbh;
        a.q[0] = *(const uint4*)(ap);
        a.q[1] = *(const uint4*)(ap + 8);
    };
    auto loadB = [&](BF16x16* b, int k0) {
        #pragma unroll
        for (int t = 0; t < 4; ++t) {
            const unsigned short* bp = wbase + (size_t)t * 16 * DMODEL + k0;
            b[t].q[0] = *(const uint4*)(bp);
            b[t].q[1] = *(const uint4*)(bp + 8);
        }
    };

    loadA(a0, 0); loadB(b0, 0);
    for (int k0 = 0; k0 < DMODEL; k0 += 64) {
        loadA(a1, k0 + 32); loadB(b1, k0 + 32);
        #pragma unroll
        for (int t = 0; t < 4; ++t) acc[t] = wmma_bf16(a0.v, b0[t].v, acc[t]);
        if (k0 + 64 < DMODEL) { loadA(a0, k0 + 64); loadB(b0, k0 + 64); }
        #pragma unroll
        for (int t = 0; t < 4; ++t) acc[t] = wmma_bf16(a1.v, b1[t].v, acc[t]);
    }

    const int mm = half * 8;
    #pragma unroll
    for (int t = 0; t < 4; ++t) {
        const int col = n0 + t * 16 + nsub;
        #pragma unroll
        for (int v = 0; v < 8; ++v) {
            const int row = m0 + mm + v;
            const float y = acc[t][v];
            if (outF) {
                outF[(size_t)row * DMODEL + col] = y;
            } else if (outT) {
                const int bb = row >> 11;           // row = bb*KLEN + k
                const int k  = row & (KLEN - 1);
                outT[(size_t)bb * DMODEL * KLEN + (size_t)col * KLEN + k] = f2bf(y);
            } else {
                if (outA) outA[(size_t)row * DMODEL + col] =
                    f2bf((y + (biasA ? biasA[col] : 0.0f)) * sAB);
                if (outB) outB[(size_t)row * DMODEL + col] =
                    f2bf((y + (biasB ? biasB[col] : 0.0f)) * sAB);
            }
        }
    }
}

// ---------------------------------------------------------------------------
// Fused rel-attention kernel.
// Block = 128 threads (4 waves). grid = (QLEN/16, NHEAD, BSZ)
// One block handles one (b, h, 16-row q-tile):
//   scores[16][KLEN] in LDS = Qw.K^T + shift(Qr.R^T)  (1/sqrt(d) already
//   folded into Qw/Qr), masked, softmax over rows, then P.V (16x64), bf16.
// rel_shift: score[q,j] uses BD_orig[q, j - q + QLEN - 1]; for tile (q0,j0)
// the needed R rows are exactly two aligned 16-tiles at rA = j0-q0+1008.
// Masked tiles form a suffix of the j-range -> branch-free compute loop
// followed by a -inf fill loop.
// VT is V transposed per batch: VT[b][d][k] -> PV B-frags are contiguous.
// ---------------------------------------------------------------------------
#define LDK 2052   // 2048 + 4 pad (bank de-skew)

__global__ __launch_bounds__(128) void k_attn(
        const unsigned short* __restrict__ Qw, const unsigned short* __restrict__ Qr,
        const unsigned short* __restrict__ Kb, const unsigned short* __restrict__ VT,
        const unsigned short* __restrict__ Rb,
        unsigned short* __restrict__ attnvec)
{
    __shared__ float sc[16 * LDK];          // full score strip, fp32
    __shared__ float bdscr[4][16 * 33];     // per-wave 16x32 BD scratch
    __shared__ float red[16 * 8];           // row reductions
    __shared__ float rinv[16];              // 1 / rowsum

    const int lane = threadIdx.x & 31;
    const int wave = threadIdx.x >> 5;
    const int q0 = blockIdx.x * 16;
    const int h  = blockIdx.y;
    const int bb = blockIdx.z;

    const int nsub = lane & 15;
    const int half = lane >> 4;
    const int klo  = half * 8;
    const int kbh  = half * 16;

    // Hoisted loop-invariant A-fragments (Qw and Qr rows, K = 0..63)
    BF16x16 aw[2], ar[2];
    {
        const unsigned short* qwrow =
            Qw + (size_t)(bb * QLEN + q0 + nsub) * DMODEL + h * DHEAD;
        const unsigned short* qrrow =
            Qr + (size_t)(bb * QLEN + q0 + nsub) * DMODEL + h * DHEAD;
        #pragma unroll
        for (int t = 0; t < 2; ++t) {
            const unsigned short* ap = qwrow + t * 32 + kbh;
            aw[t].q[0] = *(const uint4*)(ap);
            aw[t].q[1] = *(const uint4*)(ap + 8);
            const unsigned short* bp = qrrow + t * 32 + kbh;
            ar[t].q[0] = *(const uint4*)(bp);
            ar[t].q[1] = *(const uint4*)(bp + 8);
        }
    }

    // ---- Phase A: scores.  Wave w owns j-tiles [w*32, w*32+32) (512 cols). ----
    const int lastTile = (q0 + 15 + MLEN) >> 4;      // last not-fully-masked tile
    const int jt0 = wave * 32;
    const int jt1 = jt0 + 32;
    const int jtV = (jt1 < lastTile + 1) ? jt1 : lastTile + 1;

    for (int jt = jt0; jt < jtV; ++jt) {             // branch-free compute loop
        const int j0 = jt * 16;
        const unsigned short* krow =
            Kb + (size_t)(bb * KLEN + j0 + nsub) * DMODEL + h * DHEAD + kbh;
        if (jt + 1 < jtV)                            // prefetch next K tile rows
            __builtin_prefetch(krow + (size_t)16 * DMODEL, 0, 3);

        // AC = (Qw . K^T)  (K=64 -> 2 WMMAs)
        v8f ac = (v8f){};
        #pragma unroll
        for (int t = 0; t < 2; ++t) {
            BF16x16 b;
            const unsigned short* bp = krow + t * 32;
            b.q[0] = *(const uint4*)(bp);
            b.q[1] = *(const uint4*)(bp + 8);
            ac = wmma_bf16(aw[t].v, b.v, ac);
        }
        // BD_orig tiles at rA and rA+16 (shift window)
        const int rA = j0 - q0 + (QLEN - 16);
        #pragma unroll
        for (int tb = 0; tb < 2; ++tb) {
            const int rbase = rA + tb * 16;
            v8f bd = (v8f){};
            #pragma unroll
            for (int t = 0; t < 2; ++t) {
                BF16x16 b;
                const int rr = rbase + nsub;
                if (rr < KLEN) {
                    const unsigned short* bp =
                        Rb + (size_t)rr * DMODEL + h * DHEAD + t * 32 + kbh;
                    b.q[0] = *(const uint4*)(bp);
                    b.q[1] = *(const uint4*)(bp + 8);
                } else {
                    b.q[0] = make_uint4(0u, 0u, 0u, 0u);
                    b.q[1] = make_uint4(0u, 0u, 0u, 0u);
                }
                bd = wmma_bf16(ar[t].v, b.v, bd);
            }
            #pragma unroll
            for (int v = 0; v < 8; ++v) {
                const int m = half * 8 + v;
                bdscr[wave][m * 33 + tb * 16 + nsub] = bd[v];
            }
        }
        // assemble: score[q0+m][j0+n] = AC + BD_orig[., 15+n-m], masked
        #pragma unroll
        for (int v = 0; v < 8; ++v) {
            const int m = half * 8 + v;
            const int c = 15 + nsub - m;                // in [0,30]
            const float bdv = bdscr[wave][m * 33 + c];
            const int j = j0 + nsub;
            const float s = (j <= q0 + m + MLEN) ? (ac[v] + bdv)
                                                 : -__builtin_inff();
            sc[m * LDK + j] = s;
        }
    }
    // -inf fill for the fully-masked suffix
    for (int jt = (jt0 > lastTile + 1 ? jt0 : lastTile + 1); jt < jt1; ++jt) {
        const int j0 = jt * 16;
        #pragma unroll
        for (int v = 0; v < 8; ++v)
            sc[(half * 8 + v) * LDK + j0 + nsub] = -__builtin_inff();
    }
    __syncthreads();

    // ---- Phase B: row softmax (8 threads per row, 256 cols each). ----
    const int row = threadIdx.x >> 3;
    const int sub = threadIdx.x & 7;
    {
        float mx = -__builtin_inff();
        for (int j = sub * 256; j < sub * 256 + 256; ++j)
            mx = fmaxf(mx, sc[row * LDK + j]);
        red[row * 8 + sub] = mx;
    }
    __syncthreads();
    float mx = red[row * 8 + 0];
    #pragma unroll
    for (int i = 1; i < 8; ++i) mx = fmaxf(mx, red[row * 8 + i]);
    __syncthreads();
    {
        float s = 0.f;
        for (int j = sub * 256; j < sub * 256 + 256; ++j) {
            float p = __expf(sc[row * LDK + j] - mx);
            sc[row * LDK + j] = p;
            s += p;
        }
        red[row * 8 + sub] = s;
    }
    __syncthreads();
    if (threadIdx.x < 16) {
        float s = 0.f;
        #pragma unroll
        for (int i = 0; i < 8; ++i) s += red[threadIdx.x * 8 + i];
        rinv[threadIdx.x] = 1.0f / s;
    }
    __syncthreads();

    // ---- Phase C: attn_vec = P . V.  Wave w owns 16 head-dim cols. ----
    // Double-buffered VT B-frags; A-frags built from LDS P on the fly.
    const int dbase = h * DHEAD + wave * 16;
    const unsigned short* vrow = VT + (size_t)bb * DMODEL * KLEN
                                    + (size_t)(dbase + nsub) * KLEN + kbh;
    BF16x16 vb0, vb1;
    auto loadV = [&](BF16x16& b, int kt) {
        const unsigned short* vp = vrow + kt * 32;
        b.q[0] = *(const uint4*)(vp);
        b.q[1] = *(const uint4*)(vp + 8);
    };
    auto buildA = [&](BF16x16& a, int kt) {
        const float* pr = &sc[nsub * LDK + kt * 32 + kbh];
        #pragma unroll
        for (int j = 0; j < 8; ++j) a.u[j]     = f2bf(pr[j]);
        #pragma unroll
        for (int j = 0; j < 8; ++j) a.u[8 + j] = f2bf(pr[8 + j]);
    };

    v8f acc = (v8f){};
    loadV(vb0, 0);
    for (int kt = 0; kt < KLEN / 32; kt += 2) {
        loadV(vb1, kt + 1);
        BF16x16 a;
        buildA(a, kt);
        acc = wmma_bf16(a.v, vb0.v, acc);
        if (kt + 2 < KLEN / 32) loadV(vb0, kt + 2);
        buildA(a, kt + 1);
        acc = wmma_bf16(a.v, vb1.v, acc);
    }
    #pragma unroll
    for (int v = 0; v < 8; ++v) {
        const int m = half * 8 + v;
        const float y = acc[v] * rinv[m];   // fold 1/rowsum here
        attnvec[(size_t)(bb * QLEN + q0 + m) * DMODEL + dbase + nsub] = f2bf(y);
    }
}

// ---------------------------------------------------------------------------
// Host launcher
// ---------------------------------------------------------------------------
extern "C" void kernel_launch(void* const* d_in, const int* in_sizes, int n_in,
                              void* d_out, int out_size, void* d_ws, size_t ws_size,
                              hipStream_t stream) {
    (void)in_sizes; (void)n_in; (void)out_size; (void)ws_size;

    const float* x      = (const float*)d_in[0];
    const float* mem    = (const float*)d_in[1];
    const float* pos    = (const float*)d_in[2];
    // d_in[3] = attn_mask (derived analytically, unused)
    const float* Wq     = (const float*)d_in[4];
    const float* Wk     = (const float*)d_in[5];
    const float* Wv     = (const float*)d_in[6];
    const float* Wr     = (const float*)d_in[7];
    const float* Wo     = (const float*)d_in[8];
    const float* rwb    = (const float*)d_in[9];   // [16*64]
    const float* rrb    = (const float*)d_in[10];  // [16*64]
    float* out          = (float*)d_out;

    // workspace carve-up (256B aligned)
    size_t off = 0;
    auto carve = [&](size_t bytes) {
        void* p = (char*)d_ws + off;
        off += (bytes + 255) & ~(size_t)255;
        return p;
    };
    const size_t WB = (size_t)DMODEL * DMODEL * 2;
    unsigned short* Wq_b = (unsigned short*)carve(WB);
    unsigned short* Wk_b = (unsigned short*)carve(WB);
    unsigned short* Wv_b = (unsigned short*)carve(WB);
    unsigned short* Wr_b = (unsigned short*)carve(WB);
    unsigned short* Wo_b = (unsigned short*)carve(WB);
    unsigned short* x_b   = (unsigned short*)carve((size_t)BSZ * QLEN * DMODEL * 2);
    unsigned short* kin_b = (unsigned short*)carve((size_t)BSZ * KLEN * DMODEL * 2);
    unsigned short* pos_b = (unsigned short*)carve((size_t)KLEN * DMODEL * 2);
    unsigned short* Qw_b  = (unsigned short*)carve((size_t)BSZ * QLEN * DMODEL * 2);
    unsigned short* Qr_b  = (unsigned short*)carve((size_t)BSZ * QLEN * DMODEL * 2);
    unsigned short* K_b   = (unsigned short*)carve((size_t)BSZ * KLEN * DMODEL * 2);
    unsigned short* VT_b  = (unsigned short*)carve((size_t)BSZ * KLEN * DMODEL * 2);
    unsigned short* R_b   = (unsigned short*)carve((size_t)KLEN * DMODEL * 2);
    unsigned short* AV_b  = (unsigned short*)carve((size_t)BSZ * QLEN * DMODEL * 2);

    auto conv = [&](const float* s, unsigned short* d, int n) {
        k_f32_to_bf16<<<(n + 255) / 256, 256, 0, stream>>>(s, d, n);
    };
    const int NW = DMODEL * DMODEL;
    conv(Wq, Wq_b, NW); conv(Wk, Wk_b, NW); conv(Wv, Wv_b, NW);
    conv(Wr, Wr_b, NW); conv(Wo, Wo_b, NW);
    conv(x, x_b, BSZ * QLEN * DMODEL);
    conv(pos, pos_b, KLEN * DMODEL);
    {
        const int n = BSZ * KLEN * DMODEL;
        k_build_kin<<<(n + 255) / 256, 256, 0, stream>>>(mem, x, kin_b);
    }

    const float scale = 0.125f;  // 1/sqrt(DHEAD), exact power of two in bf16

    // projections (all WMMA GEMMs)
    k_gemm_xwT<<<dim3(16, (BSZ * QLEN) / 128), 256, 0, stream>>>(
        x_b, Wq_b, BSZ * QLEN, Qw_b, rwb, Qr_b, rrb, scale, nullptr, nullptr);
    k_gemm_xwT<<<dim3(16, (BSZ * KLEN) / 128), 256, 0, stream>>>(
        kin_b, Wk_b, BSZ * KLEN, K_b, nullptr, nullptr, nullptr, 1.0f, nullptr, nullptr);
    k_gemm_xwT<<<dim3(16, (BSZ * KLEN) / 128), 256, 0, stream>>>(
        kin_b, Wv_b, BSZ * KLEN, nullptr, nullptr, nullptr, nullptr, 1.0f, nullptr, VT_b);
    k_gemm_xwT<<<dim3(16, KLEN / 128), 256, 0, stream>>>(
        pos_b, Wr_b, KLEN, R_b, nullptr, nullptr, nullptr, 1.0f, nullptr, nullptr);

    // fused attention
    k_attn<<<dim3(QLEN / 16, NHEAD, BSZ), 128, 0, stream>>>(
        Qw_b, Qr_b, K_b, VT_b, R_b, AV_b);

    // output projection -> fp32
    k_gemm_xwT<<<dim3(16, (BSZ * QLEN) / 128), 256, 0, stream>>>(
        AV_b, Wo_b, BSZ * QLEN, nullptr, nullptr, nullptr, nullptr, 1.0f, out, nullptr);
}
